// SpectralLinear_773094114037
// MI455X (gfx1250) — matrix operand order repaired and checked by reading, not compiled, not run on hardware
//
#include <hip/hip_runtime.h>

// ---------------------------------------------------------------------------
// SpectralLinear on MI455X (gfx1250, wave32)
//   spec scatter -> 2D inverse FFT (rows then cols) -> W, then
//   out = x @ W^T + bias  via v_wmma_f32_16x16x32_bf16 with 3-term hi/lo
//   bf16 split for ~fp32 accuracy at bf16 matrix-engine rate.
//   GEMM: 64x128 block tile, 32x32 wave tile (2x2 accums, 12 WMMA / k-step),
//   double-buffered LDS filled with GLOBAL_LOAD_ASYNC_TO_LDS_B128 (ASYNCcnt).
// ---------------------------------------------------------------------------

#define OUT_F 4096
#define IN_F  4096
#define MROWS 2048            // 4 * 512
#define FFT_N 4096
#define LOG2N 12

#define BM 64
#define BN 128
#define BK 32

typedef __attribute__((ext_vector_type(16))) __bf16 v16bf;
typedef __attribute__((ext_vector_type(8)))  __bf16 v8bf;
typedef __attribute__((ext_vector_type(8)))  float  v8f;
typedef __attribute__((ext_vector_type(4)))  int    v4i;

// ------------- async global->LDS copy (CDNA5 path, guarded) ----------------
#if defined(__has_builtin)
#if __has_builtin(__builtin_amdgcn_global_load_async_to_lds_b128)
#define HAVE_ASYNC_LDS 1
#endif
#endif

__device__ __forceinline__ void cp_b128(void* lds, const void* g) {
#ifdef HAVE_ASYNC_LDS
    // builtin expects pointer-to-int4 in AS1 (global) / AS3 (LDS)
    __builtin_amdgcn_global_load_async_to_lds_b128(
        (__attribute__((address_space(1))) v4i*)g,
        (__attribute__((address_space(3))) v4i*)lds, 0, 0);
#else
    *(uint4*)lds = *(const uint4*)g;
#endif
}

__device__ __forceinline__ void wait_async() {
#ifdef HAVE_ASYNC_LDS
#if __has_builtin(__builtin_amdgcn_s_wait_asynccnt)
    __builtin_amdgcn_s_wait_asynccnt(0);
#else
    asm volatile("s_wait_asynccnt 0x0" ::: "memory");
#endif
#endif
}

// ------------------------- zero the dense spectrum -------------------------
__global__ __launch_bounds__(256) void zero_spec_kernel(float4* __restrict__ p) {
    // 4096*4096 complex64 = 8,388,608 float4; grid sized exactly
    size_t i = (size_t)blockIdx.x * 256 + threadIdx.x;
    p[i] = make_float4(0.f, 0.f, 0.f, 0.f);
}

// ------------------------- scatter sparse coefficients ---------------------
__global__ __launch_bounds__(256) void scatter_kernel(float2* __restrict__ spec,
                                                      const float* __restrict__ sr,
                                                      const float* __restrict__ si,
                                                      const int* __restrict__ idx,
                                                      int nnz) {
    int i = blockIdx.x * 256 + threadIdx.x;
    if (i < nnz) {
        unsigned u = (unsigned)idx[i];
        if (u < (unsigned)(OUT_F * IN_F))
            spec[u] = make_float2(sr[i], si[i]);
    }
}

// ------------------------- radix-2 inverse FFT core (in LDS) ---------------
__device__ __forceinline__ void ifft4096_lds(float2* sh, int tid) {
    // bit-reversal permutation (pairs are disjoint; only owner of smaller idx swaps)
    #pragma unroll
    for (int j = 0; j < 16; ++j) {
        int i = tid + 256 * j;
        int r = (int)(__brev((unsigned)i) >> (32 - LOG2N));
        if (r > i) { float2 t = sh[i]; sh[i] = sh[r]; sh[r] = t; }
    }
    __syncthreads();
    // 12 radix-2 stages, inverse twiddle sign (+)
    for (int s = 1; s <= LOG2N; ++s) {
        const int hm = 1 << (s - 1);
        const float step = 6.283185307179586f / (float)(1 << s);
        #pragma unroll 2
        for (int j = 0; j < 8; ++j) {
            int t  = tid + 256 * j;              // 0..2047 butterflies
            int k  = t & (hm - 1);
            int i1 = ((t >> (s - 1)) << s) + k;
            int i2 = i1 + hm;
            float wsn, wcs;
            __sincosf(step * (float)k, &wsn, &wcs);
            float2 u = sh[i1];
            float2 v = sh[i2];
            float vr = v.x * wcs - v.y * wsn;
            float vi = v.x * wsn + v.y * wcs;
            sh[i1] = make_float2(u.x + vr, u.y + vi);
            sh[i2] = make_float2(u.x - vr, u.y - vi);
        }
        __syncthreads();
    }
}

// ------------------------- pass 1: IFFT along rows (contiguous) ------------
__global__ __launch_bounds__(256) void ifft_rows_kernel(float2* __restrict__ spec) {
    __shared__ float2 sh[FFT_N];
    const int tid = threadIdx.x;
    float2* base = spec + (size_t)blockIdx.x * FFT_N;
    #pragma unroll
    for (int j = 0; j < 16; ++j) sh[tid + 256 * j] = base[tid + 256 * j];
    __syncthreads();
    ifft4096_lds(sh, tid);
    const float inv = 1.0f / (float)FFT_N;
    #pragma unroll
    for (int j = 0; j < 16; ++j) {
        int i = tid + 256 * j;
        base[i] = make_float2(sh[i].x * inv, sh[i].y * inv);
    }
}

// ---------- pass 2: IFFT along cols; emit hi/lo bf16 split of W ------------
__global__ __launch_bounds__(256) void ifft_cols_split_kernel(const float2* __restrict__ spec,
                                                              __bf16* __restrict__ wh,
                                                              __bf16* __restrict__ wl) {
    __shared__ float2 sh[FFT_N];
    const int tid = threadIdx.x;
    const int col = blockIdx.x;
    #pragma unroll
    for (int j = 0; j < 16; ++j) {
        int r = tid + 256 * j;
        sh[r] = spec[(size_t)r * IN_F + col];
    }
    __syncthreads();
    ifft4096_lds(sh, tid);
    const float inv = 1.0f / (float)FFT_N;     // second 1/N factor of ifft2
    #pragma unroll
    for (int j = 0; j < 16; ++j) {
        int r = tid + 256 * j;                 // r = output row index (o)
        float wv = sh[r].x * inv;              // real part of IFFT2
        __bf16 h  = (__bf16)wv;
        __bf16 l  = (__bf16)(wv - (float)h);
        size_t o  = (size_t)r * IN_F + col;    // W row-major [o][i]
        wh[o] = h;
        wl[o] = l;
    }
}

// ------------------------- split x into hi/lo bf16 -------------------------
__global__ __launch_bounds__(256) void split_x_kernel(const float* __restrict__ x,
                                                      __bf16* __restrict__ xh,
                                                      __bf16* __restrict__ xl) {
    size_t i = (size_t)blockIdx.x * 256 + threadIdx.x;   // exactly MROWS*IN_F threads
    float v = x[i];
    __bf16 h = (__bf16)v;
    xh[i] = h;
    xl[i] = (__bf16)(v - (float)h);
}

// ------------------------- WMMA GEMM: out = X @ W^T + bias -----------------
// Block: 256 threads (8 waves, 2(M) x 4(N)). Block tile BM=64 x BN=128.
// Wave tile 32x32 = 2x2 accumulators of 16x16 -> 12 WMMAs per BK=32 chunk.
// Double-buffered LDS, filled by async global->LDS (ASYNCcnt pipeline).
__global__ __launch_bounds__(256) void gemm_wmma_kernel(const __bf16* __restrict__ Xh,
                                                        const __bf16* __restrict__ Xl,
                                                        const __bf16* __restrict__ Wh,
                                                        const __bf16* __restrict__ Wl,
                                                        const float* __restrict__ bias,
                                                        float* __restrict__ out) {
    // pad: row stride 80B keeps 16B alignment for b128 LDS access
    __shared__ __align__(16) __bf16 sXh[2][BM][40];
    __shared__ __align__(16) __bf16 sXl[2][BM][40];
    __shared__ __align__(16) __bf16 sWh[2][BN][40];
    __shared__ __align__(16) __bf16 sWl[2][BN][40];

    const int tid  = threadIdx.x;
    const int lane = tid & 31;
    const int wid  = tid >> 5;       // 0..7
    const int half = lane >> 4;      // 0/1 (lane group)
    const int lm   = lane & 15;

    const int m0 = blockIdx.y * BM;
    const int n0 = blockIdx.x * BN;
    const int wm = (wid & 1) * 32;   // wave M offset inside block tile
    const int wn = (wid >> 1) * 32;  // wave N offset inside block tile

    // cooperative fill coordinates (one b128 = 8 bf16 per load)
    const int xr = tid >> 2, xk = (tid & 3) * 8;   // X: 64 rows x 32 k, 1 b128/thread/matrix
    const int wr = tid >> 1, wk = (tid & 1) * 16;  // W: 128 rows x 32 k, 2 b128/thread/matrix

    const __bf16* gXh = Xh + (size_t)(m0 + xr) * IN_F + xk;
    const __bf16* gXl = Xl + (size_t)(m0 + xr) * IN_F + xk;
    const __bf16* gWh = Wh + (size_t)(n0 + wr) * IN_F + wk;
    const __bf16* gWl = Wl + (size_t)(n0 + wr) * IN_F + wk;

    v8f c[2][2] = {};

    auto fill = [&](int buf, int k0) {
        cp_b128(&sXh[buf][xr][xk],      gXh + k0);
        cp_b128(&sXl[buf][xr][xk],      gXl + k0);
        cp_b128(&sWh[buf][wr][wk],      gWh + k0);
        cp_b128(&sWh[buf][wr][wk + 8],  gWh + k0 + 8);
        cp_b128(&sWl[buf][wr][wk],      gWl + k0);
        cp_b128(&sWl[buf][wr][wk + 8],  gWl + k0 + 8);
    };

    // prologue: stage first K-chunk
    fill(0, 0);
    wait_async();
    __syncthreads();

    int cur = 0;
    for (int k0 = 0; k0 < IN_F; k0 += BK) {
        const int nxt = cur ^ 1;
        if (k0 + BK < IN_F) fill(nxt, k0 + BK);            // async fill of next tile
        if (k0 + 2 * BK < IN_F) {                          // warm L2 two tiles ahead
            __builtin_prefetch(gWh + k0 + 2 * BK, 0, 3);
            __builtin_prefetch(gXh + k0 + 2 * BK, 0, 3);
        }

        // ---- fragment loads (ISA 7.12.2 16-bit A(16x32)/B(32x16) layouts) ----
        v16bf ah[2], al[2], bh[2], bl[2];
        #pragma unroll
        for (int t = 0; t < 2; ++t) {
            const int row = wm + 16 * t + lm;
            ((v8bf*)&ah[t])[0] = *(const v8bf*)&sXh[cur][row][8 * half];
            ((v8bf*)&ah[t])[1] = *(const v8bf*)&sXh[cur][row][16 + 8 * half];
            ((v8bf*)&al[t])[0] = *(const v8bf*)&sXl[cur][row][8 * half];
            ((v8bf*)&al[t])[1] = *(const v8bf*)&sXl[cur][row][16 + 8 * half];
        }
        #pragma unroll
        for (int u = 0; u < 2; ++u) {
            const int row = wn + 16 * u + lm;
            ((v8bf*)&bh[u])[0] = *(const v8bf*)&sWh[cur][row][8 * half];
            ((v8bf*)&bh[u])[1] = *(const v8bf*)&sWh[cur][row][16 + 8 * half];
            ((v8bf*)&bl[u])[0] = *(const v8bf*)&sWl[cur][row][8 * half];
            ((v8bf*)&bl[u])[1] = *(const v8bf*)&sWl[cur][row][16 + 8 * half];
        }

        // ---- 12 WMMAs: 3-term split xh*wh + xh*wl + xl*wh on 2x2 tiles ----
        #pragma unroll
        for (int t = 0; t < 2; ++t) {
            #pragma unroll
            for (int u = 0; u < 2; ++u) {
                c[t][u] = __builtin_amdgcn_wmma_f32_16x16x32_bf16(false, ah[t], false, bh[u], (short)0, c[t][u], false, false);
                c[t][u] = __builtin_amdgcn_wmma_f32_16x16x32_bf16(false, ah[t], false, bl[u], (short)0, c[t][u], false, false);
                c[t][u] = __builtin_amdgcn_wmma_f32_16x16x32_bf16(false, al[t], false, bh[u], (short)0, c[t][u], false, false);
            }
        }

        wait_async();      // next tile has landed in LDS
        __syncthreads();   // everyone done reading cur / sees nxt
        cur = nxt;
    }

    // ---- epilogue: C/D layout: VGPR r holds (M = r + 8*half, N = lm) ----
    #pragma unroll
    for (int t = 0; t < 2; ++t) {
        #pragma unroll
        for (int u = 0; u < 2; ++u) {
            const int n  = n0 + wn + 16 * u + lm;
            const int mb = m0 + wm + 16 * t + (half << 3);
            const float bv = bias[n];
            #pragma unroll
            for (int r = 0; r < 8; ++r)
                out[(size_t)(mb + r) * OUT_F + n] = c[t][u][r] + bv;
        }
    }
}

// ---------------------------------------------------------------------------
extern "C" void kernel_launch(void* const* d_in, const int* in_sizes, int n_in,
                              void* d_out, int out_size, void* d_ws, size_t ws_size,
                              hipStream_t stream) {
    const float* x    = (const float*)d_in[0];
    const float* sr   = (const float*)d_in[1];
    const float* si   = (const float*)d_in[2];
    const int*   sidx = (const int*)d_in[3];
    const float* bias = (const float*)d_in[4];
    float*       out  = (float*)d_out;
    const int nnz = in_sizes[1];

    // workspace layout
    char* ws = (char*)d_ws;
    float2* spec = (float2*)ws;                                 // 128 MB
    __bf16* Wh   = (__bf16*)(ws + (size_t)134217728);           //  32 MB
    __bf16* Wl   = (__bf16*)(ws + (size_t)167772160);           //  32 MB
    __bf16* Xh   = (__bf16*)(ws + (size_t)201326592);           //  16 MB
    __bf16* Xl   = (__bf16*)(ws + (size_t)218103808);           //  16 MB

    // 1) zero dense spectrum (8,388,608 float4)
    zero_spec_kernel<<<32768, 256, 0, stream>>>((float4*)spec);
    // 2) scatter sparse coefficients
    scatter_kernel<<<(nnz + 255) / 256, 256, 0, stream>>>(spec, sr, si, sidx, nnz);
    // 3) inverse FFT along rows (in place, scaled 1/N)
    ifft_rows_kernel<<<FFT_N, 256, 0, stream>>>(spec);
    // 4) inverse FFT along cols; emit W as hi/lo bf16 split (scaled 1/N)
    ifft_cols_split_kernel<<<IN_F, 256, 0, stream>>>(spec, Wh, Wl);
    // 5) split activations into hi/lo bf16 (2048*4096 / 256 = 32768 blocks)
    split_x_kernel<<<32768, 256, 0, stream>>>(x, Xh, Xl);
    // 6) WMMA GEMM: out[2048,4096] = X @ W^T + bias
    dim3 grid(OUT_F / BN, MROWS / BM);
    gemm_wmma_kernel<<<grid, 256, 0, stream>>>(Xh, Xl, Wh, Wl, bias, out);
}
